// MOLELinear_87411174408787
// MI455X (gfx1250) — matrix operand order repaired and checked by reading, not compiled, not run on hardware
//
#include <hip/hip_runtime.h>

#define B_GROUPS 64
#define S_TOK    8192
#define E_EXP    8
#define I_FEAT   128
#define O_FEAT   128

typedef _Float16 v16h __attribute__((ext_vector_type(16)));
typedef float    v8f  __attribute__((ext_vector_type(8)));
typedef _Float16 h2   __attribute__((ext_vector_type(2)));
typedef _Float16 h4   __attribute__((ext_vector_type(4)));

typedef unsigned int u32x4 __attribute__((ext_vector_type(4)));
typedef int          i32x4 __attribute__((ext_vector_type(4)));
typedef int          i32x8 __attribute__((ext_vector_type(8)));

#if defined(__has_builtin)
#if __has_builtin(__builtin_amdgcn_tensor_load_to_lds)
#define HAVE_TDM 1
#endif
#endif

// ---------------------------------------------------------------------------
// Phase 1: mix expert weights/biases per group.
//   wmix[b,o,i] = sum_e coeff[b,e] * W[e,o,i]   (stored f16)
//   bmix[b,o]   = sum_e coeff[b,e] * bias[e,o]  (stored f32)
// ---------------------------------------------------------------------------
__global__ __launch_bounds__(256) void mole_mix_kernel(
    const float* __restrict__ coeff,   // [B,E]
    const float* __restrict__ wexp,    // [E,O,I]
    const float* __restrict__ bexp,    // [E,O]
    _Float16* __restrict__ wmix,       // [B,O,I] f16
    float* __restrict__ bmix)          // [B,O]   f32
{
    const int b   = blockIdx.x;
    const int tid = threadIdx.x;

    float c[E_EXP];
#pragma unroll
    for (int e = 0; e < E_EXP; ++e) c[e] = coeff[b * E_EXP + e];

    const int WN = O_FEAT * I_FEAT;   // 16384
    for (int idx = tid; idx < WN; idx += 256) {
        float s = 0.f;
#pragma unroll
        for (int e = 0; e < E_EXP; ++e) s += c[e] * wexp[e * WN + idx];
        wmix[(size_t)b * WN + idx] = (_Float16)s;
    }
    if (tid < O_FEAT) {
        float s = 0.f;
#pragma unroll
        for (int e = 0; e < E_EXP; ++e) s += c[e] * bexp[e * O_FEAT + tid];
        bmix[b * O_FEAT + tid] = s;
    }
}

// ---------------------------------------------------------------------------
// TDM: load one 128x128 f16 tile (row stride 128 halves in memory) into LDS
// with 4-DWORD padding after every 64 DWORDs -> LDS row stride 136 halves.
// D# layout per CDNA5 ISA §8 (groups 0/1; groups 2/3 zero for a 2D tensor).
// ---------------------------------------------------------------------------
#if HAVE_TDM
__device__ __forceinline__ void tdm_load_w_tile(const _Float16* gsrc, void* lds_dst)
{
    unsigned long long ga = (unsigned long long)(uintptr_t)gsrc;
    unsigned int lds_off  = (unsigned int)(uintptr_t)lds_dst; // low 32b of flat shared addr

    u32x4 g0;
    g0[0] = 1u;                                          // count=1, user D#
    g0[1] = lds_off;                                     // lds_addr
    g0[2] = (unsigned int)(ga & 0xffffffffu);            // global_addr[31:0]
    g0[3] = (unsigned int)((ga >> 32) & 0x01ffffffu)     // global_addr[56:32]
          | (2u << 30);                                  // type = 2 ("image")

    i32x8 g1;
    g1[0] = (1 << 16)        // data_size = 2 bytes
          | (1 << 20)        // pad_enable
          | (5 << 22)        // pad_interval: 64 DWORDs (one 128-half row)
          | (3 << 25);       // pad_amount:   4 DWORDs (8 halves)
    g1[1] = (128 << 16);     // tensor_dim0 = 128 (bits 63:48)
    g1[2] = (128 << 16);     // tensor_dim0 hi = 0; tensor_dim1 = 128 (bits 95:80)
    g1[3] = (128 << 16);     // tensor_dim1 hi = 0; tile_dim0 = 128 (bits 127:112)
    g1[4] = 128;             // tile_dim1 = 128; tile_dim2 = 0
    g1[5] = 128;             // tensor_dim0_stride = 128 (bits 191:160)
    g1[6] = 0;               // stride hi / tensor_dim1_stride lo
    g1[7] = 0;

    i32x4 z4 = {0, 0, 0, 0};
#if __clang_major__ >= 23
    i32x8 z8 = {0, 0, 0, 0, 0, 0, 0, 0};
    __builtin_amdgcn_tensor_load_to_lds(g0, g1, z4, z4, z8, 0);
#else
    __builtin_amdgcn_tensor_load_to_lds(g0, g1, z4, z4, 0);
#endif
}
#endif

// ---------------------------------------------------------------------------
// Phase 2: per-group GEMM  out[b,s,:] = x[b,s,:] @ mixed_w[b]^T + mixed_b[b]
// Block: 256 threads (8 waves), 64 S-rows x 128 O-cols per block.
// Wave w: M-tile = (w & 3) (16 rows), N-half = (w >> 2) (4 N-tiles of 16).
// WMMA: v_wmma_f32_16x16x32_f16, f32 accumulate, K-loop of 4 steps.
// W tile arrives via TDM (tensor_load_to_lds) overlapped with the VALU
// f32->f16 conversion of the x tile.
// ---------------------------------------------------------------------------
__global__ __launch_bounds__(256) void mole_gemm_kernel(
    const float* __restrict__ x,         // [B*S, I] f32
    const _Float16* __restrict__ wmix,   // [B,O,I]  f16
    const float* __restrict__ bmix,      // [B,O]    f32
    float* __restrict__ out)             // [B*S, O] f32
{
    __shared__ _Float16 xs[64][136];     // x tile, f16, padded stride
    __shared__ _Float16 wsh[128][136];   // W tile  (row = o, col = i)

    const int b     = blockIdx.y;
    const int stile = blockIdx.x;        // 0..127
    const int tid   = threadIdx.x;

#if HAVE_TDM
    // ---- kick off the async W-tile DMA first (one issue, wave 0)
    if (tid < 32)
        tdm_load_w_tile(wmix + (size_t)b * (O_FEAT * I_FEAT), &wsh[0][0]);
#endif

    // ---- load x tile (64 rows x 128 f32) -> convert -> f16 LDS (overlaps TDM)
    const float4* xsrc =
        reinterpret_cast<const float4*>(x + ((size_t)b * S_TOK + stile * 64) * I_FEAT);
#pragma unroll
    for (int it = 0; it < 8; ++it) {
        int idx = tid + it * 256;        // 0..2047 over 64 rows x 32 float4
        int row = idx >> 5;
        int c4  = idx & 31;
        float4 f = xsrc[idx];
        h4 hv;
        hv.x = (_Float16)f.x; hv.y = (_Float16)f.y;
        hv.z = (_Float16)f.z; hv.w = (_Float16)f.w;
        *reinterpret_cast<h4*>(&xs[row][c4 * 4]) = hv;
    }
    // speculative prefetch of the next x tile (global_prefetch_b8)
    if ((b * 128 + stile) < (B_GROUPS * 128 - 1))
        __builtin_prefetch(xsrc + 2048, 0, 0);

#if HAVE_TDM
    if (tid < 32)
        __builtin_amdgcn_s_wait_tensorcnt(0);   // TDM completion fence (wave 0)
#else
    // ---- fallback: manual W-tile copy (128 x 128 f16) -> LDS (b128 copies)
    {
        const uint4* wsrc =
            reinterpret_cast<const uint4*>(wmix + (size_t)b * (O_FEAT * I_FEAT));
#pragma unroll
        for (int it = 0; it < 8; ++it) {
            int idx = tid + it * 256;    // 0..2047 over 128 rows x 8 uint4
            int row = idx >> 3;
            int c8  = idx & 7;
            *reinterpret_cast<uint4*>(&wsh[row][c8 * 8]) = wsrc[idx];
        }
    }
#endif
    __syncthreads();

    const int lane  = tid & 31;
    const int wave  = tid >> 5;
    const int mtile = wave & 3;          // 0..3 : 16-row tile
    const int nhalf = wave >> 2;         // 0..1 : 64-col half
    const int l16   = lane & 15;
    const int half  = lane >> 4;

    // accumulators seeded with bias (bias depends on column only)
    v8f acc[4];
#pragma unroll
    for (int nt = 0; nt < 4; ++nt) {
        int col  = (nhalf * 4 + nt) * 16 + l16;
        float bv = bmix[b * O_FEAT + col];
        v8f a;
#pragma unroll
        for (int r = 0; r < 8; ++r) a[r] = bv;
        acc[nt] = a;
    }

    // A fragment source row (row = token), B fragment source row (row = o)
    const _Float16* xrow = &xs[mtile * 16 + l16][0];

#pragma unroll
    for (int k0 = 0; k0 < 128; k0 += 32) {
        // ---- A fragment: 16x32 f16; lane holds row (l16), K striped:
        //  VGPR v<4 : K = k0 + half*8 + 2v ; v>=4 : K = k0 + 16 + half*8 + 2(v-4)
        v16h afr;
#pragma unroll
        for (int v = 0; v < 4; ++v) {
            h2 p = *reinterpret_cast<const h2*>(xrow + (k0 + half * 8 + 2 * v));
            afr[2 * v]     = p.x;
            afr[2 * v + 1] = p.y;
        }
#pragma unroll
        for (int v = 4; v < 8; ++v) {
            h2 p = *reinterpret_cast<const h2*>(xrow + (k0 + 16 + half * 8 + 2 * (v - 4)));
            afr[2 * v]     = p.x;
            afr[2 * v + 1] = p.y;
        }

#pragma unroll
        for (int nt = 0; nt < 4; ++nt) {
            // ---- B fragment: 32x16 f16; lane holds column o; K striped as A:
            //      v<4 : K = k0 + half*8 + 2v ; v>=4 : K = k0 + 16 + half*8 + 2(v-4)
            const _Float16* wrow = &wsh[(nhalf * 4 + nt) * 16 + l16][0];
            v16h bfr;
#pragma unroll
            for (int v = 0; v < 8; ++v) {
                h2 p = *reinterpret_cast<const h2*>(wrow + (k0 + half * 8 + 2 * v + (v >= 4 ? 8 : 0)));
                bfr[2 * v]     = p.x;
                bfr[2 * v + 1] = p.y;
            }
            acc[nt] = __builtin_amdgcn_wmma_f32_16x16x32_f16(
                false, afr, false, bfr, (short)0, acc[nt], false, false);
        }
    }

    // ---- store: acc VGPR r, lane -> (row = r + 8*half, col = tile*16 + l16)
    float* obase = out + ((size_t)b * S_TOK + stile * 64 + mtile * 16) * O_FEAT;
#pragma unroll
    for (int nt = 0; nt < 4; ++nt) {
        int col = (nhalf * 4 + nt) * 16 + l16;
#pragma unroll
        for (int r = 0; r < 8; ++r) {
            obase[(r + 8 * half) * O_FEAT + col] = acc[nt][r];
        }
    }
}

// ---------------------------------------------------------------------------
extern "C" void kernel_launch(void* const* d_in, const int* in_sizes, int n_in,
                              void* d_out, int out_size, void* d_ws, size_t ws_size,
                              hipStream_t stream) {
    (void)in_sizes; (void)n_in; (void)out_size; (void)ws_size;

    const float* x     = (const float*)d_in[0];   // [B*S, I]
    const float* coeff = (const float*)d_in[1];   // [B, E]
    const float* wexp  = (const float*)d_in[2];   // [E, O, I]
    const float* bexp  = (const float*)d_in[3];   // [E, O]
    // d_in[4] = sizes (int64), all equal S -> unused (static shapes)
    float* out = (float*)d_out;

    char* ws = (char*)d_ws;
    _Float16* wmix = (_Float16*)ws;                                        // 2 MiB
    float*    bmix = (float*)(ws + (size_t)B_GROUPS * O_FEAT * I_FEAT * 2); // 32 KiB

    mole_mix_kernel<<<B_GROUPS, 256, 0, stream>>>(coeff, wexp, bexp, wmix, bmix);
    mole_gemm_kernel<<<dim3(S_TOK / 64, B_GROUPS), 256, 0, stream>>>(x, wmix, bmix, out);
}
